// AttentionWordRNN_55774445306179
// MI455X (gfx1250) — compile-verified, matrix operand order checked by reading, not run on hardware
//
#include <hip/hip_runtime.h>
#include <hip/hip_bf16.h>
#include <math.h>

// ---- WMMA vector types (CDNA5 wave32) ----
typedef _Float16 v16h __attribute__((ext_vector_type(16)));
typedef _Float16 v8h  __attribute__((ext_vector_type(8)));
typedef _Float16 v4h  __attribute__((ext_vector_type(4)));
typedef float    v8f  __attribute__((ext_vector_type(8)));

#define S_LEN 2048
#define BATCH 32
#define VDIM  128
#define HDIM  128
#define G3H   384      // 3*H
#define D2H   256      // 2*H
#define MROWS (S_LEN*BATCH)   // 65536 rows

// ---- CDNA5 async global->LDS (ASYNCcnt) with safe fallback ----
#if defined(__AMDGCN__) && __has_builtin(__builtin_amdgcn_global_load_async_to_lds_b128)
#define HAVE_ASYNC_LDS 1
// builtin prototype: (AS1 int4* src, AS3 int4* dst, imm offset, imm cpol)
typedef int async_v4i __attribute__((vector_size(16)));
typedef __attribute__((address_space(1))) async_v4i* g_v4i_p;
typedef __attribute__((address_space(3))) async_v4i* l_v4i_p;
#else
#define HAVE_ASYNC_LDS 0
#endif

#if __has_builtin(__builtin_amdgcn_s_wait_asynccnt)
#define WAIT_ASYNC(n) __builtin_amdgcn_s_wait_asynccnt(n)
#elif defined(__AMDGCN__)
#define WAIT_ASYNC(n) asm volatile("s_wait_asynccnt %0" ::"i"(n) : "memory")
#else
#define WAIT_ASYNC(n)
#endif

__device__ __forceinline__ void async_copy16(const _Float16* gsrc, _Float16* ldst) {
#if HAVE_ASYNC_LDS
  __builtin_amdgcn_global_load_async_to_lds_b128((g_v4i_p)gsrc, (l_v4i_p)ldst, 0, 0);
#else
  *(uint4*)ldst = *(const uint4*)gsrc;
#endif
}

__device__ __forceinline__ float fast_sigmoid(float x) {
  return 1.f / (1.f + __expf(-x));
}
__device__ __forceinline__ float fast_tanh(float x) {
  return 1.f - 2.f / (__expf(2.f * x) + 1.f);
}

// A-operand loader: row-major f16 row, 16x32 tile K-pattern per ISA:
//   lane l: m = l&15, hi = l>>4; elem e -> k = (e>>3)*16 + hi*8 + (e&7)
// => two contiguous 16B loads per lane.
__device__ __forceinline__ v16h load_a16(const _Float16* row, int kbase, int hi) {
  const v8h* p0 = (const v8h*)(row + kbase + hi * 8);
  const v8h* p1 = (const v8h*)(row + kbase + 16 + hi * 8);
  v8h lo = *p0, hh = *p1;
  v16h a;
#pragma unroll
  for (int i = 0; i < 8; ++i) { a[i] = lo[i]; a[i + 8] = hh[i]; }
  return a;
}

// B-operand loader from pre-packed layout: tile = 512 halfs, 32B/lane contiguous.
__device__ __forceinline__ v16h load_bpk(const _Float16* tile, int lane) {
  const v8h* p = (const v8h*)(tile + lane * 16);
  v8h x0 = p[0], x1 = p[1];
  v16h b;
#pragma unroll
  for (int i = 0; i < 8; ++i) { b[i] = x0[i]; b[i + 8] = x1[i]; }
  return b;
}

__device__ __forceinline__ v8f wmma16(v16h a, v16h b, v8f c) {
  return __builtin_amdgcn_wmma_f32_16x16x32_f16(false, a, false, b, (short)0, c,
                                                false, false);
}

// ---------- embedding gather + f32->f16 ----------
__global__ void k_gather(const int* __restrict__ embed,
                         const float* __restrict__ table,
                         _Float16* __restrict__ X16) {
  int t = blockIdx.x * blockDim.x + threadIdx.x;       // 2,097,152 threads
  size_t flat = (size_t)t * 4;                          // 4 elems each
  int row = (int)(flat >> 7);                           // / V
  int v   = (int)(flat & (VDIM - 1));
  int s = row >> 5, b = row & 31;                       // row = s*B + b
  int tok = embed[b * S_LEN + s];                       // embed is [B,S]
  float4 val = *(const float4*)(table + (size_t)tok * VDIM + v);
  v4h o;
  o[0] = (_Float16)val.x; o[1] = (_Float16)val.y;
  o[2] = (_Float16)val.z; o[3] = (_Float16)val.w;
  *(v4h*)(X16 + flat) = o;
}

// ---------- pack weight matrix into WMMA B-operand layout ----------
// trans=1: B[k][n] = W[n*K + k]  (gi/gh: B = W^T)
// trans=0: B[k][n] = W[k*N + n]  (W_word: out @ W_word)
__global__ void k_pack(const float* __restrict__ W, _Float16* __restrict__ dst,
                       int N, int K, int trans) {
  int idx = blockIdx.x * blockDim.x + threadIdx.x;
  if (idx >= N * K) return;
  int elem = idx & 15;
  int lane = (idx >> 4) & 31;
  int tile = idx >> 9;
  int nT = N >> 4;
  int nTile = tile % nT;
  int kt = tile / nT;
  int hi = lane >> 4;
  int n = nTile * 16 + (lane & 15);
  int k = kt * 32 + ((elem >> 3) * 16 + hi * 8 + (elem & 7));
  float v = trans ? W[n * K + k] : W[k * N + n];
  dst[idx] = (_Float16)v;
}

// ---------- Gi = X @ Wih^T + bih  (65536 x 384, K=128), f16 out ----------
__global__ void k_gemm_gi(const _Float16* __restrict__ X,
                          const _Float16* __restrict__ Bp,
                          const float* __restrict__ bih,
                          _Float16* __restrict__ Gi) {
  int wave = threadIdx.x >> 5, lane = threadIdx.x & 31;
  int hi = lane >> 4, lo = lane & 15;
  int mTile = blockIdx.x * 8 + wave;    // 4096 m-tiles
  int nTile = blockIdx.y;               // 24 n-tiles
  const _Float16* row = X + (size_t)(mTile * 16 + lo) * VDIM;
  v8f acc = {};
#pragma unroll
  for (int kt = 0; kt < 4; ++kt) {
    v16h a = load_a16(row, kt * 32, hi);
    v16h b = load_bpk(Bp + (size_t)(kt * 24 + nTile) * 512, lane);
    acc = wmma16(a, b, acc);
  }
  int n = nTile * 16 + lo;
  float bias = bih[n];
  size_t base = (size_t)(mTile * 16 + hi * 8) * G3H + n;
#pragma unroll
  for (int r = 0; r < 8; ++r)
    Gi[base + (size_t)r * G3H] = (_Float16)(acc[r] + bias);
}

// ---------- serial GRU scan, one workgroup per direction ----------
// LDS: Whh packed 96KB | ghbuf f32 48KB | h16 8KB | gi double-buffer 48KB = 200KB
__global__ void k_scan(const _Float16* __restrict__ Gi_f,
                       const _Float16* __restrict__ Gi_b,
                       const _Float16* __restrict__ Whh_fp,
                       const _Float16* __restrict__ Whh_bp,
                       const float* __restrict__ bhh_f,
                       const float* __restrict__ bhh_b,
                       _Float16* __restrict__ out16) {
  extern __shared__ char smem[];
  _Float16* whh = (_Float16*)smem;                 // 49152 halfs
  float*    ghb = (float*)(smem + 98304);          // 32*384 f32
  _Float16* h16 = (_Float16*)(smem + 147456);      // 32*128 f16
  _Float16* gib = (_Float16*)(smem + 155648);      // 2 x 12288 halfs

  int dir = blockIdx.x;
  const _Float16* Gi   = dir ? Gi_b   : Gi_f;
  const _Float16* Whhp = dir ? Whh_bp : Whh_fp;
  const float*    bhh  = dir ? bhh_b  : bhh_f;

  int tid = threadIdx.x;
  { // stage packed Whh into LDS once (96KB)
    const uint4* src = (const uint4*)Whhp;
    uint4* dst = (uint4*)whh;
    for (int i = tid; i < 6144; i += 256) dst[i] = src[i];
  }
  for (int i = tid; i < BATCH * HDIM; i += 256) h16[i] = (_Float16)0.f;

  int wave = tid >> 5, lane = tid & 31;
  int hi = lane >> 4, lo = lane & 15;

  // per-thread fixed column j; hidden state lives in registers
  int j = tid & 127;
  float br = bhh[j], bz = bhh[HDIM + j], bn = bhh[2 * HDIM + j];
  float hreg[16];
#pragma unroll
  for (int k = 0; k < 16; ++k) hreg[k] = 0.f;

  // async prefetch of gi slab for t=0 into buffer 0 (6 issues per wave)
  {
    int s0 = dir ? (S_LEN - 1) : 0;
    const _Float16* src = Gi + (size_t)s0 * BATCH * G3H;
    for (int c = tid; c < 1536; c += 256)
      async_copy16(src + c * 8, gib + c * 8);
  }
  __syncthreads();

  for (int t = 0; t < S_LEN; ++t) {
    // issue async prefetch for step t+1 into the other buffer
    {
      int tn = (t + 1 < S_LEN) ? (t + 1) : t;
      int sn = dir ? (S_LEN - 1 - tn) : tn;
      const _Float16* src = Gi + (size_t)sn * BATCH * G3H;
      _Float16* dstb = gib + ((t + 1) & 1) * 12288;
      for (int c = tid; c < 1536; c += 256)
        async_copy16(src + c * 8, dstb + c * 8);
    }

    // gh = h @ Whh^T : A fragments loaded once (both mTiles), 3 nTiles/wave
    v16h a0[4], a1[4];
#pragma unroll
    for (int kt = 0; kt < 4; ++kt) {
      a0[kt] = load_a16(h16 + lo * HDIM, kt * 32, hi);
      a1[kt] = load_a16(h16 + (16 + lo) * HDIM, kt * 32, hi);
    }
#pragma unroll
    for (int nt = 0; nt < 3; ++nt) {
      int nTile = wave * 3 + nt;
      v8f acc0 = {}, acc1 = {};
#pragma unroll
      for (int kt = 0; kt < 4; ++kt) {
        v16h b = load_bpk(whh + (kt * 24 + nTile) * 512, lane);
        acc0 = wmma16(a0[kt], b, acc0);
        acc1 = wmma16(a1[kt], b, acc1);
      }
      int n = nTile * 16 + lo;
#pragma unroll
      for (int r = 0; r < 8; ++r) {
        ghb[(hi * 8 + r) * G3H + n]        = acc0[r];
        ghb[(16 + hi * 8 + r) * G3H + n]   = acc1[r];
      }
    }

    // own 6 async loads for this step's slab must have landed
    WAIT_ASYNC(6);
    __syncthreads();

    // gates + state update from LDS gi buffer
    const _Float16* gis = gib + (t & 1) * 12288;
    int s = dir ? (S_LEN - 1 - t) : t;
#pragma unroll
    for (int k = 0; k < 16; ++k) {
      int i = tid + (k << 8);
      int b = i >> 7;
      const _Float16* g = gis + b * G3H;
      float ir  = (float)g[j];
      float iz  = (float)g[HDIM + j];
      float inn = (float)g[2 * HDIM + j];
      float hr = ghb[b * G3H + j]            + br;
      float hz = ghb[b * G3H + HDIM + j]     + bz;
      float hn = ghb[b * G3H + 2 * HDIM + j] + bn;
      float rr = fast_sigmoid(ir + hr);
      float zz = fast_sigmoid(iz + hz);
      float nn = fast_tanh(inn + rr * hn);
      float hNew = (1.f - zz) * nn + zz * hreg[k];
      hreg[k] = hNew;
      h16[i] = (_Float16)hNew;
      out16[((size_t)s * BATCH + b) * D2H + dir * HDIM + j] = (_Float16)hNew;
    }
    __syncthreads();
  }
}

// ---------- attn[s,b] = tanh(out @ W_word + b_word) . w_proj ----------
__global__ void k_attn(const _Float16* __restrict__ out16,
                       const _Float16* __restrict__ Wwp,
                       const float* __restrict__ bword,
                       const float* __restrict__ wproj,
                       float* __restrict__ attn) {
  int wave = threadIdx.x >> 5, lane = threadIdx.x & 31;
  int hi = lane >> 4, lo = lane & 15;
  int mTile = blockIdx.x * 8 + wave;   // 4096 m-tiles
  const _Float16* row = out16 + (size_t)(mTile * 16 + lo) * D2H;
  v16h areg[8];
#pragma unroll
  for (int kt = 0; kt < 8; ++kt) areg[kt] = load_a16(row, kt * 32, hi);
  float p[8];
#pragma unroll
  for (int r = 0; r < 8; ++r) p[r] = 0.f;
  for (int nTile = 0; nTile < 16; ++nTile) {
    v8f acc = {};
#pragma unroll
    for (int kt = 0; kt < 8; ++kt) {
      v16h b = load_bpk(Wwp + (size_t)(kt * 16 + nTile) * 512, lane);
      acc = wmma16(areg[kt], b, acc);
    }
    int n = nTile * 16 + lo;
    float bw = bword[n], wp = wproj[n];
#pragma unroll
    for (int r = 0; r < 8; ++r)
      p[r] += fast_tanh(acc[r] + bw) * wp;
  }
  // reduce over the 16 lanes (N dim) of each half-wave
#pragma unroll
  for (int r = 0; r < 8; ++r) {
    float v = p[r];
#pragma unroll
    for (int m = 1; m <= 8; m <<= 1) v += __shfl_xor(v, m, 32);
    if (lo == 0) attn[mTile * 16 + hi * 8 + r] = v;
  }
}

// ---------- softmax over batch (B=32 == wave32) per position s ----------
__global__ void k_softmax(const float* __restrict__ attn,
                          float* __restrict__ attnN) {
  int wave = threadIdx.x >> 5, lane = threadIdx.x & 31;
  int s = blockIdx.x * 8 + wave;
  float v = attn[s * BATCH + lane];
  float mx = v;
#pragma unroll
  for (int m = 1; m <= 16; m <<= 1) mx = fmaxf(mx, __shfl_xor(mx, m, 32));
  float e = __expf(v - mx);
  float sum = e;
#pragma unroll
  for (int m = 1; m <= 16; m <<= 1) sum += __shfl_xor(sum, m, 32);
  attnN[s * BATCH + lane] = e / sum;
}

// ---------- pooled[b,:] += sum_{s in chunk} attnN[s,b]*out[s,b,:] ----------
__global__ void k_pool(const float* __restrict__ attnN,
                       const _Float16* __restrict__ out16,
                       float* __restrict__ pooled) {
  int b = blockIdx.x, chunk = blockIdx.y, d = threadIdx.x;
  int s0 = chunk * 256;
  float acc = 0.f;
  for (int s = s0; s < s0 + 256; ++s) {
    float w = attnN[s * BATCH + b];
    acc += w * (float)out16[((size_t)s * BATCH + b) * D2H + d];
  }
  atomicAdd(&pooled[b * D2H + d], acc);
}

// ---------- FC: out[b,:] = pooled[b,:] @ fc_w^T + fc_b ----------
__global__ void k_fc(const float* __restrict__ pooled,
                     const float* __restrict__ fcw,
                     const float* __restrict__ fcb,
                     float* __restrict__ out) {
  __shared__ float red0[256], red1[256];
  int b = blockIdx.x, d = threadIdx.x;
  float p = pooled[b * D2H + d];
  red0[d] = p * fcw[d];
  red1[d] = p * fcw[D2H + d];
  __syncthreads();
  for (int step = 128; step > 0; step >>= 1) {
    if (d < step) { red0[d] += red0[d + step]; red1[d] += red1[d + step]; }
    __syncthreads();
  }
  if (d == 0) {
    out[b * 2 + 0] = red0[0] + fcb[0];
    out[b * 2 + 1] = red1[0] + fcb[1];
  }
}

extern "C" void kernel_launch(void* const* d_in, const int* in_sizes, int n_in,
                              void* d_out, int out_size, void* d_ws, size_t ws_size,
                              hipStream_t stream) {
  (void)in_sizes; (void)n_in; (void)out_size; (void)ws_size;
  const int*   embed  = (const int*)d_in[0];
  const float* table  = (const float*)d_in[1];
  const float* Wih_f  = (const float*)d_in[2];
  const float* Whh_f  = (const float*)d_in[3];
  const float* bih_f  = (const float*)d_in[4];
  const float* bhh_f  = (const float*)d_in[5];
  const float* Wih_b  = (const float*)d_in[6];
  const float* Whh_b  = (const float*)d_in[7];
  const float* bih_b  = (const float*)d_in[8];
  const float* bhh_b  = (const float*)d_in[9];
  const float* W_word = (const float*)d_in[10];
  const float* b_word = (const float*)d_in[11];
  const float* w_proj = (const float*)d_in[12];
  const float* fc_w   = (const float*)d_in[13];
  const float* fc_b   = (const float*)d_in[14];
  float* outp = (float*)d_out;

  char* ws = (char*)d_ws;
  size_t off = 0;
  auto carve = [&](size_t bytes) -> char* {
    char* p = ws + off;
    off += (bytes + 255) & ~(size_t)255;
    return p;
  };
  _Float16* X16    = (_Float16*)carve((size_t)MROWS * VDIM * 2);   // 16 MB
  _Float16* Gi_f   = (_Float16*)carve((size_t)MROWS * G3H * 2);    // 48 MB
  _Float16* Gi_b   = (_Float16*)carve((size_t)MROWS * G3H * 2);    // 48 MB
  _Float16* out16  = (_Float16*)carve((size_t)MROWS * D2H * 2);    // 32 MB
  float*    attn   = (float*)carve((size_t)MROWS * 4);
  float*    attnN  = (float*)carve((size_t)MROWS * 4);
  float*    pooled = (float*)carve((size_t)BATCH * D2H * 4);
  _Float16* pWih_f = (_Float16*)carve((size_t)G3H * VDIM * 2);
  _Float16* pWih_b = (_Float16*)carve((size_t)G3H * VDIM * 2);
  _Float16* pWhh_f = (_Float16*)carve((size_t)G3H * HDIM * 2);
  _Float16* pWhh_b = (_Float16*)carve((size_t)G3H * HDIM * 2);
  _Float16* pWword = (_Float16*)carve((size_t)D2H * D2H * 2);

  // embedding gather -> f16 rows [s*B+b][V]
  k_gather<<<8192, 256, 0, stream>>>(embed, table, X16);

  // pack weights into WMMA B layout
  k_pack<<<192, 256, 0, stream>>>(Wih_f, pWih_f, G3H, VDIM, 1);
  k_pack<<<192, 256, 0, stream>>>(Wih_b, pWih_b, G3H, VDIM, 1);
  k_pack<<<192, 256, 0, stream>>>(Whh_f, pWhh_f, G3H, HDIM, 1);
  k_pack<<<192, 256, 0, stream>>>(Whh_b, pWhh_b, G3H, HDIM, 1);
  k_pack<<<256, 256, 0, stream>>>(W_word, pWword, D2H, D2H, 0);

  // input-side GEMMs (parallel part of the GRU)
  dim3 gg(512, 24);
  k_gemm_gi<<<gg, 256, 0, stream>>>(X16, pWih_f, bih_f, Gi_f);
  k_gemm_gi<<<gg, 256, 0, stream>>>(X16, pWih_b, bih_b, Gi_b);

  // serial recurrence: block 0 = forward, block 1 = backward (200KB LDS each)
  const size_t scanLds = 204800;
  (void)hipFuncSetAttribute((const void*)k_scan,
                            hipFuncAttributeMaxDynamicSharedMemorySize,
                            (int)scanLds);
  k_scan<<<2, 256, scanLds, stream>>>(Gi_f, Gi_b, pWhh_f, pWhh_b,
                                      bhh_f, bhh_b, out16);

  // attention scores, softmax over B, pooled + FC
  k_attn<<<512, 256, 0, stream>>>(out16, pWword, b_word, w_proj, attn);
  k_softmax<<<256, 256, 0, stream>>>(attn, attnN);
  (void)hipMemsetAsync(pooled, 0, (size_t)BATCH * D2H * 4, stream);
  dim3 gp(BATCH, 8);
  k_pool<<<gp, 256, 0, stream>>>(attnN, out16, pooled);
  k_fc<<<32, 256, 0, stream>>>(pooled, fc_w, fc_b, outp);
}
